// KANReasoningNetwork_9483287789847
// MI455X (gfx1250) — matrix operand order repaired and checked by reading, not compiled, not run on hardware
//
#include <hip/hip_runtime.h>
#include <hip/hip_bf16.h>
#include <stdint.h>

// ---------------------------------------------------------------------------
// KAN 3-layer network via Chebyshev-Bessel separation -> bf16 WMMA GEMM.
//   exp(-s*d) = sum_n c_n(-s) T_n(d-1),  c_n(a) = eps_n e^a I_n(a),  d=|xn-g|
//   out[b,o]  = sum_{K=(i,g,n)} T_n(z[b,i,g]) * (W[o,i,g]*c_n(-s[o,i]))
// N=16 terms; K = In*5*16. A generated on the fly in WMMA fragment layout in
// LDS (no 168MB materialization); B precomputed per layer, pre-swizzled into
// 32x16 bf16 B-fragment lane order. f32 accumulate, wave32.
// ---------------------------------------------------------------------------

typedef __attribute__((ext_vector_type(16))) __bf16 v16bf;
typedef __attribute__((ext_vector_type(8)))  float  v8f;

#define LOG2E 1.4426950408889634f
#define NCH   16                  // Chebyshev terms per (i,g)
#define NG    5                   // grid points

__device__ __forceinline__ float fast_exp2(float x) {
#if __has_builtin(__builtin_amdgcn_exp2f)
    return __builtin_amdgcn_exp2f(x);
#else
    return exp2f(x);
#endif
}
__device__ __forceinline__ float fast_tanh(float x) {
#if __has_builtin(__builtin_amdgcn_tanhf)
    return __builtin_amdgcn_tanhf(x);
#else
    return tanhf(x);
#endif
}
__device__ __forceinline__ float fast_rcp(float x) {
#if __has_builtin(__builtin_amdgcn_rcpf)
    return __builtin_amdgcn_rcpf(x);
#else
    return 1.0f / x;
#endif
}

// CDNA5 async tensor path: global -> LDS b128, tracked by ASYNCcnt.
__device__ __forceinline__ void async_g2l_b128(const float* gsrc, const float* ldst) {
    uint32_t lds_off = (uint32_t)(uintptr_t)ldst;
    uint64_t gaddr   = (uint64_t)(uintptr_t)gsrc;
    asm volatile("global_load_async_to_lds_b128 %0, %1, off"
                 :: "v"(lds_off), "v"(gaddr) : "memory");
}
__device__ __forceinline__ void wait_async0() {
#if __has_builtin(__builtin_amdgcn_s_wait_asynccnt)
    __builtin_amdgcn_s_wait_asynccnt(0);
#else
    asm volatile("s_wait_asynccnt 0" ::: "memory");
#endif
}

__device__ __forceinline__ uint32_t f2bf(float x) {  // RNE float->bf16 bits
    uint32_t u = __float_as_uint(x);
    return (u + 0x7FFFu + ((u >> 16) & 1u)) >> 16;
}

// c_n(-s), n=0..15: scaled Bessel via Miller downward recurrence (fp64).
__device__ void cheb_coeffs(float s, float* cn) {
    double a  = -(double)s;
    double m  = fabs(a);
    double mc = fmax(m, 1e-6);
    double tn[NCH + 1];
    double tkp1 = 0.0, tk = 1e-10, sum = 0.0;
    const int MSTART = 28;
    for (int k = MSTART; k >= 1; --k) {
        if (k <= NCH) tn[k] = tk;
        double tkm1 = tkp1 + (2.0 * (double)k / mc) * tk;
        sum += 2.0 * tk;
        tkp1 = tk; tk = tkm1;
    }
    tn[0] = tk; sum += tk;
    const double esc = exp(a + m);          // e^{a+m}; a>=0 -> e^{2a}, a<0 -> 1
    #pragma unroll
    for (int n = 0; n < NCH; ++n) {
        double c = tn[n] / sum * esc;       // eps_n * I_n(m) e^{-m} * e^{a+m}
        if (n) c *= 2.0;
        if (a < 0.0 && (n & 1)) c = -c;     // I_n(a) = (-1)^n I_n(|a|)
        cn[n] = (float)c;
    }
}

// ---------------------------------------------------------------------------
// Prep: Bw[(o16-tile, kblock, lane)][16 bf16] in exact B-fragment lane order.
// B order used here (32x16 bf16): VGPR j, lanes 0-15: K=2j,2j+1 / lanes 16-31: K=16+2j,..
// K-block of 32 = ig pair {2kb, 2kb+1}; lanes<16 <-> p0, lanes>=16 <-> p1.
// ---------------------------------------------------------------------------
__global__ __launch_bounds__(256)
void kan_prep_kernel(const float* __restrict__ W,   // (OUT, IN, 5)
                     const float* __restrict__ S,   // (OUT, IN)
                     uint16_t* __restrict__ Bw,
                     int IN, int OUT, int KB)       // KB = IN*5/2
{
    const int t    = blockIdx.x * 256 + threadIdx.x;
    const int lane = t & 31;
    const int tmp  = t >> 5;
    const int kb   = tmp % KB;
    const int ot16 = tmp / KB;
    if (ot16 >= (OUT >> 4)) return;

    const int p  = lane >> 4;
    const int ig = kb * 2 + p;
    const int i  = ig / NG;
    const int g  = ig - i * NG;
    const int o  = ot16 * 16 + (lane & 15);

    float cn[NCH];
    cheb_coeffs(S[(size_t)o * IN + i], cn);
    const float w = W[((size_t)o * IN + i) * NG + g];

    uint32_t u[8];
    #pragma unroll
    for (int j = 0; j < 8; ++j)
        u[j] = f2bf(w * cn[2 * j]) | (f2bf(w * cn[2 * j + 1]) << 16);

    uint16_t* dst = Bw + (((size_t)ot16 * KB + kb) * 32 + lane) * 16;
    ((uint4*)dst)[0] = make_uint4(u[0], u[1], u[2], u[3]);
    ((uint4*)dst)[1] = make_uint4(u[4], u[5], u[6], u[7]);
}

// ---------------------------------------------------------------------------
// Fused GEMM: block = 256 thr (8 waves) -> tile 32 b x 64 o (2 bt x 4 ot).
// A order used here (16x32 bf16): lanes 0-15: M, VGPR0-3: K=0..7, VGPR4-7: K=8..15;
//                      lanes 16-31: K=16..23 / K=24..31.
// ACT: 0 = relu, 1 = sigmoid.
// ---------------------------------------------------------------------------
template <int IN, int ACT>
__global__ __launch_bounds__(256)
void kan_gemm_kernel(const float* __restrict__ X,       // (B, IN)
                     const uint16_t* __restrict__ Bw,   // fragment-swizzled
                     float* __restrict__ Y,             // (B, OUT)
                     int OUT)
{
    constexpr int XS  = IN + 4;
    constexpr int KB  = IN * NG / 2;     // #32-wide K blocks
    constexpr int CH  = 8;               // K-blocks per LDS chunk

    __shared__ __attribute__((aligned(16))) float    xn_sh[32 * XS];
    __shared__ __attribute__((aligned(32))) uint32_t afrag[CH * 2 * 32 * 8]; // 16KB

    const int tid  = threadIdx.x;
    const int b0   = blockIdx.x * 32;
    const int wv   = tid >> 5;
    const int lane = tid & 31;
    const int btl  = wv & 1;             // b-subtile of this wave
    const int ot   = wv >> 1;            // o-subtile of this wave
    const int bo16 = blockIdx.y * 4 + ot;

    // ---- stage X tile (32 x IN) via async global->LDS, then v_tanh_f32 ----
    constexpr int NV = 32 * IN / 4;
    #pragma unroll
    for (int v = tid; v < NV; v += 256) {
        const int row = v / (IN / 4), c4 = v % (IN / 4);
        async_g2l_b128(X + (size_t)(b0 + row) * IN + c4 * 4,
                       &xn_sh[row * XS + c4 * 4]);
    }
    wait_async0();
    __syncthreads();
    for (int f = tid; f < 32 * IN; f += 256) {
        const int r = f / IN, c = f % IN;
        xn_sh[r * XS + c] = fast_tanh(xn_sh[r * XS + c]);
    }

    v8f acc = {};

    for (int kb0 = 0; kb0 < KB; kb0 += CH) {
        __syncthreads();   // tanh visibility (iter 0) / afrag reuse (later)

        // ---- produce A fragments: CH*64 tasks, 2 per thread ----
        #pragma unroll
        for (int tsk = 0; tsk < CH / 4; ++tsk) {
            const int task = tid + tsk * 256;
            const int kbr  = task >> 6;
            const int bt   = (task >> 5) & 1;
            const int p    = (task >> 4) & 1;
            const int b    = task & 15;
            const int ig   = (kb0 + kbr) * 2 + p;
            const int i    = ig / NG;
            const int g    = ig - i * NG;
            const float gv = -1.0f + 0.5f * (float)g;
            const float z  = fabsf(xn_sh[(bt * 16 + b) * XS + i] - gv) - 1.0f;
            float T[NCH];
            T[0] = 1.0f; T[1] = z;
            const float twoz = z + z;
            #pragma unroll
            for (int n = 2; n < NCH; ++n) T[n] = twoz * T[n - 1] - T[n - 2];
            uint32_t u[8];
            #pragma unroll
            for (int j = 0; j < 8; ++j)
                u[j] = f2bf(T[2 * j]) | (f2bf(T[2 * j + 1]) << 16);
            const int fb = (kbr * 2 + bt) * 32;
            // lane b + 16*p: n=0..15 of p
            *(uint4*)&afrag[(fb + b + p * 16) * 8]     = make_uint4(u[0], u[1], u[2], u[3]);
            *(uint4*)&afrag[(fb + b + p * 16) * 8 + 4] = make_uint4(u[4], u[5], u[6], u[7]);
        }
        __syncthreads();

        // ---- consume: CH chained WMMAs per wave ----
        #pragma unroll
        for (int kbr = 0; kbr < CH; ++kbr) {
            const v16bf av = *(const v16bf*)&afrag[((kbr * 2 + btl) * 32 + lane) * 8];
            const v16bf bv = *(const v16bf*)(Bw +
                 (((size_t)bo16 * KB + (kb0 + kbr)) * 32 + lane) * 16);
            acc = __builtin_amdgcn_wmma_f32_16x16x32_bf16(
                      false, av, false, bv, (short)0, acc, false, false);
        }
    }

    // ---- activation + store (C layout: VGPR r -> M = r + 8*(lane>=16)) ----
    const int Mb = (lane >> 4) << 3;
    const int N  = lane & 15;
    const int oc = bo16 * 16 + N;
    #pragma unroll
    for (int r = 0; r < 8; ++r) {
        float v = acc[r];
        if constexpr (ACT == 0) {
            v = fmaxf(v, 0.0f);
        } else {
            v = fast_rcp(1.0f + fast_exp2(-LOG2E * v));
        }
        Y[(size_t)(b0 + btl * 16 + Mb + r) * OUT + oc] = v;
    }
}

extern "C" void kernel_launch(void* const* d_in, const int* in_sizes, int n_in,
                              void* d_out, int out_size, void* d_ws, size_t ws_size,
                              hipStream_t stream) {
    (void)in_sizes; (void)n_in; (void)out_size; (void)ws_size;

    const float* x  = (const float*)d_in[0];
    const float* w1 = (const float*)d_in[1];
    const float* s1 = (const float*)d_in[2];
    const float* w2 = (const float*)d_in[3];
    const float* s2 = (const float*)d_in[4];
    const float* w3 = (const float*)d_in[5];
    const float* s3 = (const float*)d_in[6];
    float* out = (float*)d_out;

    constexpr int B = 2048, D_IN = 128, D_H = 256, D_H2 = 128, D_OUT = 64;
    constexpr int KB1 = D_IN * NG / 2;   // 320
    constexpr int KB2 = D_H  * NG / 2;   // 640
    constexpr int KB3 = D_H2 * NG / 2;   // 320

    // workspace layout (all offsets 256B aligned); total ~14.8 MB
    char* ws = (char*)d_ws;
    float*    h1  = (float*)ws;                 ws += (size_t)B * D_H  * 4;  // 2 MB
    float*    h2  = (float*)ws;                 ws += (size_t)B * D_H2 * 4;  // 1 MB
    uint16_t* Bw1 = (uint16_t*)ws;              ws += (size_t)(D_H  / 16) * KB1 * 32 * 32;
    uint16_t* Bw2 = (uint16_t*)ws;              ws += (size_t)(D_H2 / 16) * KB2 * 32 * 32;
    uint16_t* Bw3 = (uint16_t*)ws;

    // prep B matrices (independent; stream-ordered before their GEMMs)
    kan_prep_kernel<<<(D_H  / 16) * KB1 * 32 / 256, 256, 0, stream>>>(w1, s1, Bw1, D_IN, D_H,  KB1);
    kan_prep_kernel<<<(D_H2 / 16) * KB2 * 32 / 256, 256, 0, stream>>>(w2, s2, Bw2, D_H,  D_H2, KB2);
    kan_prep_kernel<<<(D_OUT/ 16) * KB3 * 32 / 256, 256, 0, stream>>>(w3, s3, Bw3, D_H2, D_OUT, KB3);

    // fused Chebyshev-A + WMMA GEMM layers
    kan_gemm_kernel<D_IN, 0><<<dim3(B / 32, D_H   / 64), 256, 0, stream>>>(x,  Bw1, h1, D_H);
    kan_gemm_kernel<D_H,  0><<<dim3(B / 32, D_H2  / 64), 256, 0, stream>>>(h1, Bw2, h2, D_H2);
    kan_gemm_kernel<D_H2, 1><<<dim3(B / 32, D_OUT / 64), 256, 0, stream>>>(h2, Bw3, out, D_OUT);
}
